// DoubleStreamBlock_31284541784404
// MI455X (gfx1250) — matrix operand order repaired and checked by reading, not compiled, not run on hardware
//
#include <hip/hip_runtime.h>
#include <hip/hip_bf16.h>

// ---------------- problem constants ----------------
#define L_IMG  4096
#define L_TXT  256
#define L_TOT  4352
#define DMODEL 2048
#define NHEADS 16
#define HDIM   128

#define EPI_QKV   0
#define EPI_RESID 1
#define EPI_GELU  2

typedef __bf16 bf16;
typedef bf16  v16bf __attribute__((ext_vector_type(16)));
typedef float v8f   __attribute__((ext_vector_type(8)));

union BF16Frag { v16bf v; unsigned int u[8]; };
union F32Frag  { v8f   v; float        f[8]; };

// hardware bf16 convert (RNE) -- lowers to v_cvt*bf16_f32 on gfx1250,
// instead of a manual integer round-to-nearest-even VALU sequence.
static __device__ __forceinline__ unsigned short f2bf(float f) {
  union { __bf16 h; unsigned short u; } cv;
  cv.h = (__bf16)f;
  return cv.u;
}

// LDS byte address of a __shared__ object: generic LDS pointers carry the
// aperture in the high 32 bits and the LDS offset in the low 32 bits.
static __device__ __forceinline__ unsigned lds_addr(const void* p) {
  return (unsigned)(uintptr_t)p;
}

// CDNA5 async global->LDS copy (ASYNCcnt tracked), 16B per lane.
static __device__ __forceinline__ void async_copy_b128(unsigned lds_off,
                                                       const void* gptr) {
  unsigned long long ga = (unsigned long long)(uintptr_t)gptr;
  asm volatile("global_load_async_to_lds_b128 %0, %1, off"
               :: "v"(lds_off), "v"(ga) : "memory");
}
static __device__ __forceinline__ void wait_asynccnt0() {
  asm volatile("s_wait_asynccnt 0x0" ::: "memory");
}

// ---------------- LayerNorm -> bf16 ----------------
__global__ __launch_bounds__(256)
void ln_to_bf16_kernel(const float* __restrict__ x,
                       const float* __restrict__ g,
                       const float* __restrict__ b,
                       unsigned short* __restrict__ out) {
  const int row = blockIdx.x;
  const float* xr = x + (size_t)row * DMODEL;
  float vals[8];
  float s = 0.f, s2 = 0.f;
#pragma unroll
  for (int i = 0; i < 8; ++i) {
    float v = xr[threadIdx.x + i * 256];
    vals[i] = v; s += v; s2 += v * v;
  }
#pragma unroll
  for (int off = 16; off > 0; off >>= 1) {
    s  += __shfl_xor(s,  off, 32);
    s2 += __shfl_xor(s2, off, 32);
  }
  __shared__ float red[2][8];
  const int wid = threadIdx.x >> 5, lane = threadIdx.x & 31;
  if (lane == 0) { red[0][wid] = s; red[1][wid] = s2; }
  __syncthreads();
  if (wid == 0) {
    float a = (lane < 8) ? red[0][lane] : 0.f;
    float c = (lane < 8) ? red[1][lane] : 0.f;
#pragma unroll
    for (int off = 4; off > 0; off >>= 1) { a += __shfl_xor(a, off, 32); c += __shfl_xor(c, off, 32); }
    if (lane == 0) { red[0][0] = a; red[1][0] = c; }
  }
  __syncthreads();
  const float mean = red[0][0] * (1.f / DMODEL);
  const float var  = red[1][0] * (1.f / DMODEL) - mean * mean;
  const float rstd = rsqrtf(var + 1e-5f);
#pragma unroll
  for (int i = 0; i < 8; ++i) {
    int c = threadIdx.x + i * 256;
    out[(size_t)row * DMODEL + c] = f2bf((vals[i] - mean) * rstd * g[c] + b[c]);
  }
}

// ---------------- generic bf16 WMMA GEMM, async double-buffered ----------------
// C[M,N] = A_bf16[M,K] @ W_f32->bf16[K,N]  (+bias, +epilogue)
// 256 thr (8 waves), tile 128x128, k-step 32.
// A tile: global_load_async_to_lds_b128 into ping-pong LDS buffers.
// W tile: software-pipelined through float4 regs (needs f32->bf16 convert).
__global__ __launch_bounds__(256)
void gemm_bf16_kernel(const unsigned short* __restrict__ A,
                      const float* __restrict__ W,
                      const float* __restrict__ bias,
                      int M, int N, int K, int epi,
                      unsigned short* __restrict__ qout,
                      unsigned short* __restrict__ kout,
                      unsigned short* __restrict__ vout,
                      int row_off, int use_rope,
                      const float* __restrict__ fcos,
                      const float* __restrict__ fsin,
                      const float* __restrict__ resid,
                      float* __restrict__ out_f32,
                      unsigned short* __restrict__ out_bf16) {
  __shared__ __align__(16) unsigned short Alds[2][128][40];  // [m][k] pad 40
  __shared__ __align__(16) unsigned short Wlds[2][128][40];  // [n][k] pad 40
  const int tid  = threadIdx.x;
  const int wid  = tid >> 5, lane = tid & 31;
  const int wm   = wid >> 2, wn   = wid & 3;
  const int hi   = lane >> 4, ln16 = lane & 15;
  const int bm   = blockIdx.y * 128;
  const int bn   = blockIdx.x * 128;
  const int nk   = K >> 5;

  // per-thread A-tile async-copy coordinates (2 x B128 per thread)
  const int au0 = tid * 2,     ar0 = au0 >> 2, as0 = (au0 & 3) * 8;
  const int au1 = tid * 2 + 1, ar1 = au1 >> 2, as1 = (au1 & 3) * 8;
  const unsigned short* aP0 = A + (size_t)(bm + ar0) * K + as0;
  const unsigned short* aP1 = A + (size_t)(bm + ar1) * K + as1;

  // per-thread W-tile coordinates (4 x float4 per thread)
  int wkr[4], wns[4];
  const float* wptr[4];
#pragma unroll
  for (int i = 0; i < 4; ++i) {
    int u = tid * 4 + i;
    wkr[i] = u >> 5; wns[i] = (u & 31) * 4;
    wptr[i] = W + (size_t)wkr[i] * N + bn + wns[i];
  }

  F32Frag acc[4][2];
#pragma unroll
  for (int mi = 0; mi < 4; ++mi)
#pragma unroll
    for (int ni = 0; ni < 2; ++ni)
#pragma unroll
      for (int v = 0; v < 8; ++v) acc[mi][ni].f[v] = 0.f;

  // prologue: async A(0) into buffer 0, W(0) into regs
  async_copy_b128(lds_addr(&Alds[0][ar0][as0]), aP0);
  async_copy_b128(lds_addr(&Alds[0][ar1][as1]), aP1);
  float4 wreg[4];
#pragma unroll
  for (int i = 0; i < 4; ++i) wreg[i] = *(const float4*)wptr[i];

  for (int ks = 0; ks < nk; ++ks) {
    const int p = ks & 1;
    // commit W regs (f32->bf16, transposed to [n][k])
#pragma unroll
    for (int i = 0; i < 4; ++i) {
      Wlds[p][wns[i] + 0][wkr[i]] = f2bf(wreg[i].x);
      Wlds[p][wns[i] + 1][wkr[i]] = f2bf(wreg[i].y);
      Wlds[p][wns[i] + 2][wkr[i]] = f2bf(wreg[i].z);
      Wlds[p][wns[i] + 3][wkr[i]] = f2bf(wreg[i].w);
    }
    wait_asynccnt0();      // this wave's async A(ks) slice is in LDS
    __syncthreads();       // everyone's slices + W stores visible

    if (ks + 1 < nk) {     // overlap tile ks+1 fetch with tile ks compute
      const int koff = (ks + 1) * 32;
      async_copy_b128(lds_addr(&Alds[p ^ 1][ar0][as0]), aP0 + koff);
      async_copy_b128(lds_addr(&Alds[p ^ 1][ar1][as1]), aP1 + koff);
      const size_t wo = (size_t)koff * N;
#pragma unroll
      for (int i = 0; i < 4; ++i) wreg[i] = *(const float4*)(wptr[i] + wo);
    }

    BF16Frag bfr[2];
#pragma unroll
    for (int ni = 0; ni < 2; ++ni) {
      int n = wn * 32 + ni * 16 + ln16;
#pragma unroll
      for (int v = 0; v < 8; ++v)
        bfr[ni].u[v] = *(const unsigned int*)&Wlds[p][n][(v < 4) ? (8 * hi + 2 * v) : (16 + 8 * hi + 2 * (v - 4))];
    }
#pragma unroll
    for (int mi = 0; mi < 4; ++mi) {
      BF16Frag afr;
      int m = wm * 64 + mi * 16 + ln16;
#pragma unroll
      for (int v = 0; v < 8; ++v) {
        int kk = (v < 4) ? (8 * hi + 2 * v) : (16 + 8 * hi + 2 * (v - 4));
        afr.u[v] = *(const unsigned int*)&Alds[p][m][kk];
      }
#pragma unroll
      for (int ni = 0; ni < 2; ++ni)
        acc[mi][ni].v = __builtin_amdgcn_wmma_f32_16x16x32_bf16(
            false, afr.v, false, bfr[ni].v, (short)0, acc[mi][ni].v, false, false);
    }
    __syncthreads();
  }

  // ---------------- epilogue ----------------
#pragma unroll
  for (int mi = 0; mi < 4; ++mi) {
#pragma unroll
    for (int ni = 0; ni < 2; ++ni) {
      int col = bn + wn * 32 + ni * 16 + ln16;
      float bv = bias[col];
#pragma unroll
      for (int v = 0; v < 8; ++v) {
        int row = bm + wm * 64 + mi * 16 + 8 * hi + v;
        float val = acc[mi][ni].f[v] + bv;
        if (epi == EPI_QKV) {
          int which = col >> 11;          // 0=q 1=k 2=v
          int cc = col & 2047;            // h*128 + dd
          if (use_rope && which < 2) {
            int dd = col & 127;
            val *= (fcos[(size_t)row * HDIM + dd] - fsin[(size_t)row * HDIM + dd]);
          }
          unsigned short* dst = (which == 0) ? qout : (which == 1) ? kout : vout;
          dst[(size_t)(row + row_off) * DMODEL + cc] = f2bf(val);
        } else if (epi == EPI_RESID) {
          out_f32[(size_t)row * N + col] = val + resid[(size_t)row * N + col];
        } else { // EPI_GELU, exact erf gelu
          out_bf16[(size_t)row * N + col] =
              f2bf(0.5f * val * (1.f + erff(val * 0.70710678118654752f)));
        }
      }
    }
  }
}

// ---------------- flash attention ----------------
// grid (L_TOT/64, NHEADS), 128 threads (4 waves), each wave owns 16 query rows.
// K tile copied global->LDS via async b128; V staged transposed via VGPRs.
__global__ __launch_bounds__(128)
void flash_attn_kernel(const unsigned short* __restrict__ Q,
                       const unsigned short* __restrict__ Km,
                       const unsigned short* __restrict__ Vm,
                       unsigned short* __restrict__ O) {
  __shared__ __align__(16) unsigned short Klds[32][136];   // [key][d]
  __shared__ __align__(16) unsigned short Vlds[128][34];   // [d][key] transposed
  __shared__ __align__(16) unsigned short Plds[4][16][34]; // per-wave P bounce
  const int tid  = threadIdx.x;
  const int wid  = tid >> 5, lane = tid & 31;
  const int hi   = lane >> 4, ln16 = lane & 15;
  const int head = blockIdx.y;
  const int q0   = blockIdx.x * 64;
  const int colbase = head * HDIM;
  const float scale = 0.08838834764831845f;  // 128^-0.5

  // Q fragment resident in registers: 16 rows x 128 (4 k-chunks of 32)
  unsigned int qf[4][8];
  {
    const unsigned short* qp = Q + (size_t)(q0 + wid * 16 + ln16) * DMODEL + colbase;
#pragma unroll
    for (int c = 0; c < 4; ++c)
#pragma unroll
      for (int v = 0; v < 8; ++v) {
        int kk = (v < 4) ? (8 * hi + 2 * v) : (16 + 8 * hi + 2 * (v - 4));
        qf[c][v] = *(const unsigned int*)(qp + c * 32 + kk);
      }
  }

  // per-thread tile-copy coordinates (4 units of 8 bf16)
  int ckey[4], cseg[4];
#pragma unroll
  for (int i = 0; i < 4; ++i) {
    int u = tid * 4 + i;
    ckey[i] = u >> 4; cseg[i] = (u & 15) * 8;
  }

  float mrow[8], lrow[8];
  F32Frag oacc[8];
#pragma unroll
  for (int v = 0; v < 8; ++v) { mrow[v] = -1e30f; lrow[v] = 0.f; }
#pragma unroll
  for (int nd = 0; nd < 8; ++nd)
#pragma unroll
    for (int v = 0; v < 8; ++v) oacc[nd].f[v] = 0.f;

  for (int kt = 0; kt < L_TOT; kt += 32) {
    // K tile: straight bf16 copy -> async global->LDS b128
#pragma unroll
    for (int i = 0; i < 4; ++i)
      async_copy_b128(lds_addr(&Klds[ckey[i]][cseg[i]]),
                      Km + (size_t)(kt + ckey[i]) * DMODEL + colbase + cseg[i]);
    // V tile: transpose via VGPRs
#pragma unroll
    for (int i = 0; i < 4; ++i) {
      uint4 vd = *(const uint4*)(Vm + (size_t)(kt + ckey[i]) * DMODEL + colbase + cseg[i]);
      unsigned int vw[4] = {vd.x, vd.y, vd.z, vd.w};
#pragma unroll
      for (int j = 0; j < 4; ++j) {
        Vlds[cseg[i] + 2 * j + 0][ckey[i]] = (unsigned short)(vw[j] & 0xffffu);
        Vlds[cseg[i] + 2 * j + 1][ckey[i]] = (unsigned short)(vw[j] >> 16);
      }
    }
    wait_asynccnt0();
    __syncthreads();

    // S = Q K^T  (two 16x16 tiles: keys 0-15, 16-31)
    F32Frag s0, s1;
#pragma unroll
    for (int v = 0; v < 8; ++v) { s0.f[v] = 0.f; s1.f[v] = 0.f; }
#pragma unroll
    for (int c = 0; c < 4; ++c) {
      BF16Frag a, b0, b1;
#pragma unroll
      for (int v = 0; v < 8; ++v) a.u[v] = qf[c][v];
#pragma unroll
      for (int v = 0; v < 8; ++v) {
        int kk = c * 32 + ((v < 4) ? (8 * hi + 2 * v) : (16 + 8 * hi + 2 * (v - 4)));
        b0.u[v] = *(const unsigned int*)&Klds[ln16][kk];
        b1.u[v] = *(const unsigned int*)&Klds[ln16 + 16][kk];
      }
      s0.v = __builtin_amdgcn_wmma_f32_16x16x32_bf16(false, a.v, false, b0.v, (short)0, s0.v, false, false);
      s1.v = __builtin_amdgcn_wmma_f32_16x16x32_bf16(false, a.v, false, b1.v, (short)0, s1.v, false, false);
    }

    // online softmax (butterfly reductions inside 16-lane halves match
    // the C-layout row ownership m = v + 8*hi)
    float corr[8];
#pragma unroll
    for (int v = 0; v < 8; ++v) {
      float a = s0.f[v] * scale, b = s1.f[v] * scale;
      float r = fmaxf(a, b);
      r = fmaxf(r, __shfl_xor(r, 1, 32));
      r = fmaxf(r, __shfl_xor(r, 2, 32));
      r = fmaxf(r, __shfl_xor(r, 4, 32));
      r = fmaxf(r, __shfl_xor(r, 8, 32));
      float mn = fmaxf(mrow[v], r);
      corr[v] = __expf(mrow[v] - mn);
      mrow[v] = mn;
      float p0 = __expf(a - mn), p1 = __expf(b - mn);
      s0.f[v] = p0; s1.f[v] = p1;
      float rs = p0 + p1;
      rs += __shfl_xor(rs, 1, 32);
      rs += __shfl_xor(rs, 2, 32);
      rs += __shfl_xor(rs, 4, 32);
      rs += __shfl_xor(rs, 8, 32);
      lrow[v] = lrow[v] * corr[v] + rs;
    }
#pragma unroll
    for (int nd = 0; nd < 8; ++nd)
#pragma unroll
      for (int v = 0; v < 8; ++v) oacc[nd].f[v] *= corr[v];

    // P: C-layout -> A-layout via per-wave LDS bounce (DS in-order per wave)
#pragma unroll
    for (int v = 0; v < 8; ++v) {
      int m = v + 8 * hi;
      Plds[wid][m][ln16]      = f2bf(s0.f[v]);
      Plds[wid][m][16 + ln16] = f2bf(s1.f[v]);
    }
    BF16Frag pf;
#pragma unroll
    for (int v = 0; v < 8; ++v) {
      int kk = (v < 4) ? (8 * hi + 2 * v) : (16 + 8 * hi + 2 * (v - 4));
      pf.u[v] = *(const unsigned int*)&Plds[wid][ln16][kk];
    }

    // O += P @ V  (8 d-chunks of 16)
#pragma unroll
    for (int nd = 0; nd < 8; ++nd) {
      BF16Frag bf;
#pragma unroll
      for (int v = 0; v < 8; ++v)
        bf.u[v] = *(const unsigned int*)&Vlds[nd * 16 + ln16][(v < 4) ? (8 * hi + 2 * v) : (16 + 8 * hi + 2 * (v - 4))];
      oacc[nd].v = __builtin_amdgcn_wmma_f32_16x16x32_bf16(
          false, pf.v, false, bf.v, (short)0, oacc[nd].v, false, false);
    }
    __syncthreads();
  }

  // normalize and store
#pragma unroll
  for (int nd = 0; nd < 8; ++nd)
#pragma unroll
    for (int v = 0; v < 8; ++v) {
      int row = q0 + wid * 16 + 8 * hi + v;
      int col = colbase + nd * 16 + ln16;
      O[(size_t)row * DMODEL + col] = f2bf(oacc[nd].f[v] / lrow[v]);
    }
}

// ---------------- host orchestration ----------------
extern "C" void kernel_launch(void* const* d_in, const int* in_sizes, int n_in,
                              void* d_out, int out_size, void* d_ws, size_t ws_size,
                              hipStream_t stream) {
  (void)in_sizes; (void)n_in; (void)out_size; (void)ws_size;
  const float* img  = (const float*)d_in[0];
  const float* txt  = (const float*)d_in[1];
  const float* fcos = (const float*)d_in[3];
  const float* fsin = (const float*)d_in[4];
  const float* i_ln1_g = (const float*)d_in[5];
  const float* i_ln1_b = (const float*)d_in[6];
  const float* i_qkv_w = (const float*)d_in[7];
  const float* i_qkv_b = (const float*)d_in[8];
  const float* i_proj_w= (const float*)d_in[9];
  const float* i_proj_b= (const float*)d_in[10];
  const float* i_ln2_g = (const float*)d_in[11];
  const float* i_ln2_b = (const float*)d_in[12];
  const float* i_fc1_w = (const float*)d_in[13];
  const float* i_fc1_b = (const float*)d_in[14];
  const float* i_fc2_w = (const float*)d_in[15];
  const float* i_fc2_b = (const float*)d_in[16];
  const float* t_ln1_g = (const float*)d_in[17];
  const float* t_ln1_b = (const float*)d_in[18];
  const float* t_qkv_w = (const float*)d_in[19];
  const float* t_qkv_b = (const float*)d_in[20];
  const float* t_proj_w= (const float*)d_in[21];
  const float* t_proj_b= (const float*)d_in[22];
  const float* t_ln2_g = (const float*)d_in[23];
  const float* t_ln2_b = (const float*)d_in[24];
  const float* t_fc1_w = (const float*)d_in[25];
  const float* t_fc1_b = (const float*)d_in[26];
  const float* t_fc2_w = (const float*)d_in[27];
  const float* t_fc2_b = (const float*)d_in[28];

  // workspace layout (~125 MB) with dead-buffer aliasing:
  //   [0 .. 4*SZ)   : ln1 | q | k | v  -> later reused as MLP mid [L_TOT,8192] bf16
  //   [4*SZ .. 5*SZ): attn_out bf16    -> later reused as ln2 bf16
  //   [5*SZ .. )    : h residual stream f32
  const size_t SZ = (size_t)L_TOT * DMODEL * 2;
  char* ws = (char*)d_ws;
  unsigned short* lnA  = (unsigned short*)(ws);
  unsigned short* qb   = (unsigned short*)(ws + SZ);
  unsigned short* kb   = (unsigned short*)(ws + 2 * SZ);
  unsigned short* vb   = (unsigned short*)(ws + 3 * SZ);
  unsigned short* mid  = (unsigned short*)(ws);              // alias [L_TOT,8192]
  unsigned short* attn = (unsigned short*)(ws + 4 * SZ);
  unsigned short* ln2  = attn;                               // alias
  float*          hbuf = (float*)(ws + 5 * SZ);
  float*          out  = (float*)d_out;

  // 1. LN1
  ln_to_bf16_kernel<<<L_IMG, 256, 0, stream>>>(img, i_ln1_g, i_ln1_b, lnA);
  ln_to_bf16_kernel<<<L_TXT, 256, 0, stream>>>(txt, t_ln1_g, t_ln1_b, lnA + (size_t)L_IMG * DMODEL);

  // 2. QKV (img gets (cos-sin) factor on q,k)
  gemm_bf16_kernel<<<dim3(48, 32), 256, 0, stream>>>(lnA, i_qkv_w, i_qkv_b,
      L_IMG, 3 * DMODEL, DMODEL, EPI_QKV, qb, kb, vb, 0, 1, fcos, fsin,
      nullptr, nullptr, nullptr);
  gemm_bf16_kernel<<<dim3(48, 2), 256, 0, stream>>>(lnA + (size_t)L_IMG * DMODEL, t_qkv_w, t_qkv_b,
      L_TXT, 3 * DMODEL, DMODEL, EPI_QKV, qb, kb, vb, L_IMG, 0, fcos, fsin,
      nullptr, nullptr, nullptr);

  // 3. joint attention over all 4352 tokens
  flash_attn_kernel<<<dim3(L_TOT / 64, NHEADS), 128, 0, stream>>>(qb, kb, vb, attn);

  // 4. proj + residual -> h (f32)
  gemm_bf16_kernel<<<dim3(16, 32), 256, 0, stream>>>(attn, i_proj_w, i_proj_b,
      L_IMG, DMODEL, DMODEL, EPI_RESID, nullptr, nullptr, nullptr, 0, 0, nullptr, nullptr,
      img, hbuf, nullptr);
  gemm_bf16_kernel<<<dim3(16, 2), 256, 0, stream>>>(attn + (size_t)L_IMG * DMODEL, t_proj_w, t_proj_b,
      L_TXT, DMODEL, DMODEL, EPI_RESID, nullptr, nullptr, nullptr, 0, 0, nullptr, nullptr,
      txt, hbuf + (size_t)L_IMG * DMODEL, nullptr);

  // 5. LN2 (h -> bf16), overwrites dead attn buffer
  ln_to_bf16_kernel<<<L_IMG, 256, 0, stream>>>(hbuf, i_ln2_g, i_ln2_b, ln2);
  ln_to_bf16_kernel<<<L_TXT, 256, 0, stream>>>(hbuf + (size_t)L_IMG * DMODEL, t_ln2_g, t_ln2_b,
                                               ln2 + (size_t)L_IMG * DMODEL);

  // 6. FC1 + exact GELU -> mid (overwrites dead ln1/q/k/v region)
  gemm_bf16_kernel<<<dim3(64, 32), 256, 0, stream>>>(ln2, i_fc1_w, i_fc1_b,
      L_IMG, 4 * DMODEL, DMODEL, EPI_GELU, nullptr, nullptr, nullptr, 0, 0, nullptr, nullptr,
      nullptr, nullptr, mid);
  gemm_bf16_kernel<<<dim3(64, 2), 256, 0, stream>>>(ln2 + (size_t)L_IMG * DMODEL, t_fc1_w, t_fc1_b,
      L_TXT, 4 * DMODEL, DMODEL, EPI_GELU, nullptr, nullptr, nullptr, 0, 0, nullptr, nullptr,
      nullptr, nullptr, mid + (size_t)L_IMG * 4 * DMODEL);

  // 7. FC2 + residual(h) -> d_out (img then txt, flat)
  gemm_bf16_kernel<<<dim3(16, 32), 256, 0, stream>>>(mid, i_fc2_w, i_fc2_b,
      L_IMG, DMODEL, 4 * DMODEL, EPI_RESID, nullptr, nullptr, nullptr, 0, 0, nullptr, nullptr,
      hbuf, out, nullptr);
  gemm_bf16_kernel<<<dim3(16, 2), 256, 0, stream>>>(mid + (size_t)L_IMG * 4 * DMODEL, t_fc2_w, t_fc2_b,
      L_TXT, DMODEL, 4 * DMODEL, EPI_RESID, nullptr, nullptr, nullptr, 0, 0, nullptr, nullptr,
      hbuf + (size_t)L_IMG * DMODEL, out + (size_t)L_IMG * DMODEL, nullptr);
}
